// LSTMStep_13657996001783
// MI455X (gfx1250) — compile-verified, matrix operand order checked by reading, not run on hardware
//
#include <hip/hip_runtime.h>
#include <hip/hip_bf16.h>
#include <stdint.h>

#define B_   128
#define T_   512
#define C_   1024
#define U_   1024
#define N4   4096   // 4*UNITS (gate GEMM N)
#define K2   2048   // 2*C     (gate GEMM K, also ab row stride)

typedef __attribute__((ext_vector_type(16))) __bf16 bf16x16;
typedef __attribute__((ext_vector_type(8)))  float  f32x8;

union Frag16 { bf16x16 v; unsigned u[8]; };
struct Stage { Frag16 a, b0, b1, b2, b3; };

// fp32 -> bf16 round-to-nearest-even (bit trick; inputs finite)
__device__ __forceinline__ unsigned short f2bf(float f) {
  union { float f; unsigned u; } x; x.f = f;
  unsigned r = x.u + 0x7FFFu + ((x.u >> 16) & 1u);
  return (unsigned short)(r >> 16);
}

// K-offset of VGPR v within a 16x32 bf16 A/B fragment (ISA 7.12.2):
// VGPR 0..3 -> K = hlf*8 + v*2 ; VGPR 4..7 -> +16
__device__ __forceinline__ int koff(int v, int hlf) {
  return ((v & 4) << 2) + (hlf << 3) + ((v & 3) << 1);
}

__device__ __forceinline__ void load_frag(Frag16& f, const unsigned short* p, int kk, int hlf) {
#pragma unroll
  for (int v = 0; v < 8; ++v)
    f.u[v] = *(const unsigned*)(p + kk + koff(v, hlf));
}

__device__ __forceinline__ void load_stage(Stage& s, const unsigned short* aR,
                                           const unsigned short* b0R, const unsigned short* b1R,
                                           const unsigned short* b2R, const unsigned short* b3R,
                                           int kk, int hlf) {
  load_frag(s.a,  aR,  kk, hlf);
  load_frag(s.b0, b0R, kk, hlf);
  load_frag(s.b1, b1R, kk, hlf);
  load_frag(s.b2, b2R, kk, hlf);
  load_frag(s.b3, b3R, kk, hlf);
}

__device__ __forceinline__ void mma_stage(const Stage& s, f32x8& a0, f32x8& a1, f32x8& a2, f32x8& a3) {
  a0 = __builtin_amdgcn_wmma_f32_16x16x32_bf16(false, s.a.v, false, s.b0.v, (short)0, a0, false, false);
  a1 = __builtin_amdgcn_wmma_f32_16x16x32_bf16(false, s.a.v, false, s.b1.v, (short)0, a1, false, false);
  a2 = __builtin_amdgcn_wmma_f32_16x16x32_bf16(false, s.a.v, false, s.b2.v, (short)0, a2, false, false);
  a3 = __builtin_amdgcn_wmma_f32_16x16x32_bf16(false, s.a.v, false, s.b3.v, (short)0, a3, false, false);
}

// ---------------------------------------------------------------------------
// One-time prep
// ---------------------------------------------------------------------------

// WUt[n][k] (bf16, n-major) = k<C ? W[k][n] : U[k-C][n]
__global__ void convert_WUt(const float* __restrict__ W, const float* __restrict__ Uk,
                            unsigned short* __restrict__ WUt) {
  int i = blockIdx.x * blockDim.x + threadIdx.x;   // 0 .. 4096*2048-1
  int n = i >> 11;
  int k = i & 2047;
  float v = (k < C_) ? W[(size_t)k * N4 + n] : Uk[(size_t)(k - C_) * N4 + n];
  WUt[i] = f2bf(v);
}

// Wdt[n][k] (bf16, n-major) = Wd[k][n]
__global__ void convert_Wdt(const float* __restrict__ Wd, unsigned short* __restrict__ Wdt) {
  int i = blockIdx.x * blockDim.x + threadIdx.x;   // 0 .. 1024*1024-1
  int n = i >> 10;
  int k = i & 1023;
  Wdt[i] = f2bf(Wd[(size_t)k * U_ + n]);
}

// word0 contribution: 1e-5 * colsum(W[C:2C, :])
__global__ void build_w0row(const float* __restrict__ W, float* __restrict__ w0) {
  int j = blockIdx.x * blockDim.x + threadIdx.x;   // 0..4095
  float s = 0.f;
  for (int r = 0; r < C_; ++r) s += W[(size_t)(C_ + r) * N4 + j];
  w0[j] = s * 1e-5f;
}

// ab[b][0:1024) = bf16(x[b,0,:]) ; ab[b][1024:2048) = bf16(h0)=0 ; c0 = 0 ; idx = -1
__global__ void init_state(const float* __restrict__ X, unsigned short* __restrict__ ab,
                           float* __restrict__ c, int* __restrict__ idxb) {
  int i = blockIdx.x * blockDim.x + threadIdx.x;   // 0..131071
  int b = i >> 10, u = i & 1023;
  ab[(size_t)b * K2 + u]      = f2bf(X[((size_t)b * T_) * C_ + u]);
  ab[(size_t)b * K2 + C_ + u] = 0;
  c[i] = 0.f;
  if (i < B_) idxb[i] = -1;
}

// ---------------------------------------------------------------------------
// Per-step kernels
// ---------------------------------------------------------------------------

// Z[128,4096] = ab (128x2048 bf16) @ WUt^T, WMMA bf16/f32-acc, software-pipelined.
// grid 64 x block 256 (8 waves). Wave tile = 16 rows x 64 cols (4 accumulators).
__global__ void gate_gemm(const unsigned short* __restrict__ ab,
                          const unsigned short* __restrict__ WUt,
                          float* __restrict__ Z) {
  const int lane  = threadIdx.x & 31;
  const int wave  = threadIdx.x >> 5;
  const int hlf   = lane >> 4;
  const int mtile = blockIdx.x >> 3;                          // 0..7   (128 rows)
  const int ncol  = (((blockIdx.x & 7) << 3) + wave) << 6;    // 0..4032 step 64

  const unsigned short* aR  = ab  + (size_t)((mtile << 4) + (lane & 15)) * K2;
  const unsigned short* b0R = WUt + (size_t)(ncol +      (lane & 15)) * K2;
  const unsigned short* b1R = WUt + (size_t)(ncol + 16 + (lane & 15)) * K2;
  const unsigned short* b2R = WUt + (size_t)(ncol + 32 + (lane & 15)) * K2;
  const unsigned short* b3R = WUt + (size_t)(ncol + 48 + (lane & 15)) * K2;

  f32x8 acc0 = {}, acc1 = {}, acc2 = {}, acc3 = {};
  Stage S0, S1;

  // software pipeline: loads for chunk n+1 issue before WMMAs of chunk n
  load_stage(S0, aR, b0R, b1R, b2R, b3R, 0, hlf);
#pragma unroll 1
  for (int kk = 32; kk <= K2 - 96; kk += 64) {
    load_stage(S1, aR, b0R, b1R, b2R, b3R, kk, hlf);
    __builtin_prefetch(b0R + kk + 128, 0, 1);     // global_prefetch_b8, 4 chunks ahead
    mma_stage(S0, acc0, acc1, acc2, acc3);
    load_stage(S0, aR, b0R, b1R, b2R, b3R, kk + 32, hlf);
    mma_stage(S1, acc0, acc1, acc2, acc3);
  }
  load_stage(S1, aR, b0R, b1R, b2R, b3R, K2 - 32, hlf);
  mma_stage(S0, acc0, acc1, acc2, acc3);
  mma_stage(S1, acc0, acc1, acc2, acc3);

  // D layout: lane -> col = lane&15 ; VGPR r -> row = r + 8*(lane>>4)
  float* zOut = Z + (size_t)((mtile << 4) + (hlf << 3)) * N4 + ncol + (lane & 15);
#pragma unroll
  for (int r = 0; r < 8; ++r) {
    zOut[(size_t)r * N4 +  0] = acc0[r];
    zOut[(size_t)r * N4 + 16] = acc1[r];
    zOut[(size_t)r * N4 + 32] = acc2[r];
    zOut[(size_t)r * N4 + 48] = acc3[r];
  }
}

// Gates + cell update; writes h as bf16 into ab[:,1024:] and pre-converts x_{t+1}.
__global__ void cell_update(const float* __restrict__ Z, const float* __restrict__ bias,
                            const float* __restrict__ W, const float* __restrict__ w0row,
                            const int* __restrict__ idxb, const float* __restrict__ X,
                            float* __restrict__ cbuf, unsigned short* __restrict__ ab, int t) {
  int i = blockIdx.x * blockDim.x + threadIdx.x;  // 0..131071
  int b = i >> 10, u = i & 1023;
  int pidx = idxb[b];
  const float* wrow = (pidx < 0) ? w0row : (W + (size_t)(C_ + pidx) * N4);
  const float* zb = Z + (size_t)b * N4;
  float zi = zb[u]          + bias[u]          + wrow[u];
  float zf = zb[U_ + u]     + bias[U_ + u]     + wrow[U_ + u];
  float zc = zb[2 * U_ + u] + bias[2 * U_ + u] + wrow[2 * U_ + u];
  float zo = zb[3 * U_ + u] + bias[3 * U_ + u] + wrow[3 * U_ + u];
  float ig = fminf(fmaxf(0.2f * zi + 0.5f, 0.f), 1.f);   // keras hard_sigmoid
  float fg = fminf(fmaxf(0.2f * zf + 0.5f, 0.f), 1.f);
  float og = fminf(fmaxf(0.2f * zo + 0.5f, 0.f), 1.f);
  float cn = fg * cbuf[i] + ig * tanhf(zc);
  float hn = og * tanhf(cn);
  cbuf[i] = cn;
  ab[(size_t)b * K2 + C_ + u] = f2bf(hn);                 // h for next GEMMs
  if (t + 1 < T_)                                         // stage x_{t+1} as bf16
    ab[(size_t)b * K2 + u] = f2bf(X[((size_t)b * T_ + t + 1) * C_ + u]);
}

// logits = h @ Wd + bd, then per-row argmax (softmax monotone -> skipped).
// grid 8 x block 512 (16 waves). WG owns 16 rows x 1024 cols; wave owns 16x64.
__global__ void logits_argmax(const unsigned short* __restrict__ ab,
                              const unsigned short* __restrict__ Wdt,
                              const float* __restrict__ bd, int* __restrict__ idxb,
                              float* __restrict__ out, int t) {
  __shared__ float sl[16 * 1024];                 // 64 KB of the 320 KB WGP LDS
  const int lane = threadIdx.x & 31;
  const int wave = threadIdx.x >> 5;              // 0..15
  const int hlf  = lane >> 4;
  const int m0   = blockIdx.x << 4;
  const unsigned short* aR  = ab + (size_t)(m0 + (lane & 15)) * K2 + C_;  // h half
  const int nb = (wave << 6) + (lane & 15);
  const unsigned short* b0R = Wdt + (size_t)(nb)      * U_;
  const unsigned short* b1R = Wdt + (size_t)(nb + 16) * U_;
  const unsigned short* b2R = Wdt + (size_t)(nb + 32) * U_;
  const unsigned short* b3R = Wdt + (size_t)(nb + 48) * U_;

  f32x8 acc0 = {}, acc1 = {}, acc2 = {}, acc3 = {};
  Stage S0, S1;

  load_stage(S0, aR, b0R, b1R, b2R, b3R, 0, hlf);
#pragma unroll 1
  for (int kk = 32; kk <= U_ - 96; kk += 64) {
    load_stage(S1, aR, b0R, b1R, b2R, b3R, kk, hlf);
    mma_stage(S0, acc0, acc1, acc2, acc3);
    load_stage(S0, aR, b0R, b1R, b2R, b3R, kk + 32, hlf);
    mma_stage(S1, acc0, acc1, acc2, acc3);
  }
  load_stage(S1, aR, b0R, b1R, b2R, b3R, U_ - 32, hlf);
  mma_stage(S0, acc0, acc1, acc2, acc3);
  mma_stage(S1, acc0, acc1, acc2, acc3);

  // stage logits (+bias) to LDS: row r+8*hlf, col per lane
  {
    int colb = (wave << 6) + (lane & 15);
#pragma unroll
    for (int r = 0; r < 8; ++r) {
      int row = r + (hlf << 3);
      sl[row * 1024 + colb +  0] = acc0[r] + bd[colb +  0];
      sl[row * 1024 + colb + 16] = acc1[r] + bd[colb + 16];
      sl[row * 1024 + colb + 32] = acc2[r] + bd[colb + 32];
      sl[row * 1024 + colb + 48] = acc3[r] + bd[colb + 48];
    }
  }
  __syncthreads();

  // one wave per row: strided scan + shuffle reduce (first-index tie-break)
  int row = wave;
  float bestv = -3.402823466e38f;
  int bestj = 0;
  for (int j = lane; j < 1024; j += 32) {
    float v = sl[row * 1024 + j];
    if (v > bestv) { bestv = v; bestj = j; }
  }
#pragma unroll
  for (int off = 16; off > 0; off >>= 1) {
    float ov = __shfl_down(bestv, off);
    int   oj = __shfl_down(bestj, off);
    if (ov > bestv || (ov == bestv && oj < bestj)) { bestv = ov; bestj = oj; }
  }
  if (lane == 0) {
    int b = m0 + row;
    idxb[b] = bestj;
    out[(size_t)b * T_ + t] = (float)bestj;
  }
}

// ---------------------------------------------------------------------------
extern "C" void kernel_launch(void* const* d_in, const int* in_sizes, int n_in,
                              void* d_out, int out_size, void* d_ws, size_t ws_size,
                              hipStream_t stream) {
  (void)in_sizes; (void)n_in; (void)out_size; (void)ws_size;
  const float* X    = (const float*)d_in[0];  // [B,T,C]
  const float* W    = (const float*)d_in[1];  // [2C,4U]
  const float* Uk   = (const float*)d_in[2];  // [U,4U]
  const float* bias = (const float*)d_in[3];  // [4U]
  const float* Wd   = (const float*)d_in[4];  // [U,C]
  const float* bd   = (const float*)d_in[5];  // [C]
  float* out = (float*)d_out;                 // [B,T] indices as float

  char* ws = (char*)d_ws;                                   // ~21.1 MiB used
  unsigned short* WUt = (unsigned short*)(ws);              // 16 MiB  bf16 [4096][2048]
  unsigned short* Wdt = (unsigned short*)(ws + (16u << 20));// 2 MiB   bf16 [1024][1024]
  float* Z     = (float*)(ws + (18u << 20));                // 2 MiB   fp32 [128][4096]
  unsigned short* ab = (unsigned short*)(ws + (20u << 20)); // 512 KiB bf16 [128][2048] = [x_t | h]
  float* cbuf  = (float*)(ws + (20u << 20) + (512u << 10)); // 512 KiB fp32 [128][1024]
  float* w0row = (float*)(ws + (21u << 20));                // 16 KiB
  int*   idxb  = (int*)  (ws + (21u << 20) + (64u << 10));  // 512 B

  convert_WUt <<<32768, 256, 0, stream>>>(W, Uk, WUt);
  convert_Wdt <<<4096,  256, 0, stream>>>(Wd, Wdt);
  build_w0row <<<16,    256, 0, stream>>>(W, w0row);
  init_state  <<<512,   256, 0, stream>>>(X, ab, cbuf, idxb);

  for (int t = 0; t < T_; ++t) {
    gate_gemm     <<<64,  256, 0, stream>>>(ab, WUt, Z);
    cell_update   <<<512, 256, 0, stream>>>(Z, bias, W, w0row, idxb, X, cbuf, ab, t);
    logits_argmax <<<8,   512, 0, stream>>>(ab, Wdt, bd, idxb, out, t);
  }
}
